// AttentionModule_33492154974223
// MI455X (gfx1250) — compile-verified, matrix operand order checked by reading, not run on hardware
//
#include <hip/hip_runtime.h>
#include <hip/hip_bf16.h>
#include <math.h>

// ---------------------------------------------------------------------------
// Shapes fixed by the reference: H=128, C=1024, O=2.
// ---------------------------------------------------------------------------
#define HDIM 128
#define CDIM 1024
#define ODIM 2
#define EPS 1e-5f

typedef __attribute__((ext_vector_type(16))) __bf16 v16bf;
typedef __attribute__((ext_vector_type(16))) unsigned short v16us;
typedef __attribute__((ext_vector_type(8))) float v8f;

union FragU {
    v16us u;
    v16bf b;
};

// round-to-nearest-even f32 -> bf16 (bit pattern in ushort)
__device__ __forceinline__ unsigned short f2bf(float f) {
    unsigned int u = __float_as_uint(f);
    unsigned int r = u + 0x7FFFu + ((u >> 16) & 1u);
    return (unsigned short)(r >> 16);
}
__device__ __forceinline__ float bf2f(unsigned short h) {
    return __uint_as_float(((unsigned int)h) << 16);
}

// ---------------------------------------------------------------------------
// Kernel 0: inclusive prefix sum of repeat_counts (tiny, single thread).
// ---------------------------------------------------------------------------
__global__ void scan_kernel(const int* __restrict__ rc, int* __restrict__ csum, int Bn) {
    if (blockIdx.x == 0 && threadIdx.x == 0) {
        int s = 0;
        for (int i = 0; i < Bn; ++i) { s += rc[i]; csum[i] = s; }
    }
}

// ---------------------------------------------------------------------------
// Kernel 1: per-context attention over the codebook + per-context precompute
//   scores = cv @ K^T / sqrt(H); w = softmax(scores)  -> out_aw [B,C]
//   ao = w @ V                                         [H]
//   rt = relu(right_W @ ao + right_b)                  [H]
//   M[b,o,i] = sum_j bil_W[o,i,j] * rt[j]              -> Mws [B,O,H]
// One 256-thread block per context.
// ---------------------------------------------------------------------------
__global__ __launch_bounds__(256) void attn_kernel(
    const float* __restrict__ ctxv, const float* __restrict__ Kmat,
    const float* __restrict__ Vmat, const float* __restrict__ rightW,
    const float* __restrict__ rightB, const float* __restrict__ bilW,
    float* __restrict__ Mws, float* __restrict__ out_aw)
{
    const int b = blockIdx.x;
    const int t = threadIdx.x;

    __shared__ float cv[HDIM];
    __shared__ float sc[CDIM];
    __shared__ float red[256];
    __shared__ float ao[HDIM];
    __shared__ float rt[HDIM];

    if (t < HDIM) cv[t] = ctxv[(size_t)b * HDIM + t];
    __syncthreads();

    const float inv_scale = 0.088388347648318447f; // 1/sqrt(128)

    // scores
    for (int c = t; c < CDIM; c += 256) {
        const float* Kr = Kmat + (size_t)c * HDIM;
        float s = 0.0f;
        #pragma unroll 8
        for (int i = 0; i < HDIM; ++i) s += cv[i] * Kr[i];
        sc[c] = s * inv_scale;
    }
    __syncthreads();

    // block max
    float m = -INFINITY;
    for (int c = t; c < CDIM; c += 256) m = fmaxf(m, sc[c]);
    red[t] = m;
    __syncthreads();
    for (int s = 128; s > 0; s >>= 1) {
        if (t < s) red[t] = fmaxf(red[t], red[t + s]);
        __syncthreads();
    }
    const float mx = red[0];
    __syncthreads();

    // exp + sum
    float ps = 0.0f;
    for (int c = t; c < CDIM; c += 256) {
        float e = expf(sc[c] - mx);
        sc[c] = e;
        ps += e;
    }
    red[t] = ps;
    __syncthreads();
    for (int s = 128; s > 0; s >>= 1) {
        if (t < s) red[t] += red[t + s];
        __syncthreads();
    }
    const float inv_sum = 1.0f / red[0];
    __syncthreads();

    // normalize, write attn_weights output
    for (int c = t; c < CDIM; c += 256) {
        float w = sc[c] * inv_sum;
        sc[c] = w;
        out_aw[(size_t)b * CDIM + c] = w;
    }
    __syncthreads();

    // attn_output: ao[h] = sum_c w[c] * V[c,h]   (lanes over h -> coalesced)
    if (t < HDIM) {
        float a = 0.0f;
        for (int c = 0; c < CDIM; ++c) a += sc[c] * Vmat[(size_t)c * HDIM + t];
        ao[t] = a;
    }
    __syncthreads();

    // right branch (per-context!)
    if (t < HDIM) {
        const float* Wr = rightW + (size_t)t * HDIM;
        float r = 0.0f;
        #pragma unroll 8
        for (int i = 0; i < HDIM; ++i) r += Wr[i] * ao[i];
        rt[t] = fmaxf(r + rightB[t], 0.0f);
    }
    __syncthreads();

    // M[b,o,i] = bil_W[o,i,:] . rt   (256 threads = O*H exactly)
    {
        const int o = t >> 7;        // 0..1
        const int i = t & (HDIM - 1);
        const float* Br = bilW + ((size_t)o * HDIM + i) * HDIM;
        float mm = 0.0f;
        #pragma unroll 8
        for (int j = 0; j < HDIM; ++j) mm += Br[j] * rt[j];
        Mws[((size_t)b * ODIM + o) * HDIM + i] = mm;
    }
}

// ---------------------------------------------------------------------------
// Kernel 2: fused token path.
//   64 tokens/block, 128 threads (4 waves). Each wave owns one 16-row tile.
//   LN(X) -> bf16 LDS; left = relu(Xn @ left_W^T + b) via v_wmma bf16 (f32 acc);
//   logits[n,o] = left[n,:] . M[b(n),o,:] + bil_b; log_softmax over O=2.
// ---------------------------------------------------------------------------
#define TB 64

__global__ __launch_bounds__(128) void token_kernel(
    const float* __restrict__ X, const float* __restrict__ leftW,
    const float* __restrict__ leftB, const float* __restrict__ bilB,
    const float* __restrict__ Mws, const int* __restrict__ csum,
    int Bn, int Nn, float* __restrict__ out_lp)
{
    __shared__ unsigned short xn[TB][HDIM];   // Xn bf16, later reused for relu(left) bf16
    __shared__ unsigned short wl[HDIM][HDIM]; // left_W bf16, [j][i]
    __shared__ int ctxid[TB];
    __shared__ float lg[TB][ODIM];

    const int n0 = blockIdx.x * TB;
    const int t  = threadIdx.x;

    // hint-prefetch the next block's X tile (gfx1250 global_prefetch path)
    if ((n0 + TB) * HDIM < Nn * HDIM) {
        __builtin_prefetch(X + (size_t)(n0 + TB) * HDIM, 0, 1);
    }

    // load left_W -> bf16 LDS (coalesced)
    for (int idx = t; idx < HDIM * HDIM; idx += 128) {
        wl[idx >> 7][idx & (HDIM - 1)] = f2bf(leftW[idx]);
    }

    // LayerNorm (torch-style: unbiased std, divide by std+eps), one row/thread
    if (t < TB) {
        const int n = n0 + t;
        if (n < Nn) {
            const float* Xr = X + (size_t)n * HDIM;
            float s = 0.0f, s2 = 0.0f;
            #pragma unroll 8
            for (int i = 0; i < HDIM; ++i) {
                float v = Xr[i];
                s += v;
                s2 += v * v;
            }
            const float mu  = s * (1.0f / HDIM);
            float var = (s2 - (float)HDIM * mu * mu) * (1.0f / (HDIM - 1));
            var = fmaxf(var, 0.0f);
            const float inv = 1.0f / (sqrtf(var) + EPS);
            #pragma unroll 8
            for (int i = 0; i < HDIM; ++i) xn[t][i] = f2bf((Xr[i] - mu) * inv);
            // context id: first b with csum[b] > n
            int lo = 0, hi = Bn - 1;
            while (lo < hi) {
                int mid = (lo + hi) >> 1;
                if (csum[mid] > n) hi = mid; else lo = mid + 1;
            }
            ctxid[t] = lo;
        } else {
            for (int i = 0; i < HDIM; ++i) xn[t][i] = 0;
            ctxid[t] = 0;
        }
    }
    __syncthreads();

    // ---- WMMA: left = Xn @ left_W^T, wave w -> rows [16w, 16w+16) ----
    const int lane = t & 31;
    const int w    = t >> 5;           // wave id 0..3
    const int m0   = w * 16;
    const int mrow = lane & 15;        // M (or N) within tile
    const int kh   = lane >> 4;        // which K-half this lane holds

    v8f acc[8];
    #pragma unroll
    for (int ct = 0; ct < 8; ++ct) acc[ct] = (v8f){0,0,0,0,0,0,0,0};

    #pragma unroll
    for (int kc = 0; kc < 4; ++kc) {
        const int k0 = kc * 32;
        FragU a;
        #pragma unroll
        for (int e = 0; e < 8; ++e) a.u[e]     = xn[m0 + mrow][k0 + kh * 8 + e];
        #pragma unroll
        for (int e = 0; e < 8; ++e) a.u[8 + e] = xn[m0 + mrow][k0 + 16 + kh * 8 + e];

        #pragma unroll
        for (int ct = 0; ct < 8; ++ct) {
            const int j = ct * 16 + mrow;  // output column owned by this lane
            FragU bf;
            #pragma unroll
            for (int e = 0; e < 8; ++e) bf.u[e]     = wl[j][k0 + kh * 8 + e];
            #pragma unroll
            for (int e = 0; e < 8; ++e) bf.u[8 + e] = wl[j][k0 + 16 + kh * 8 + e];
            acc[ct] = __builtin_amdgcn_wmma_f32_16x16x32_bf16(
                false, a.b, false, bf.b, (short)0, acc[ct], false, false);
        }
    }
    __syncthreads();

    // relu(left + bias) -> reuse xn as bf16 left tile
    #pragma unroll
    for (int ct = 0; ct < 8; ++ct) {
        const int col = ct * 16 + (lane & 15);
        const float bias = leftB[col];
        #pragma unroll
        for (int r = 0; r < 8; ++r) {
            const int row = m0 + r + 8 * kh;
            xn[row][col] = f2bf(fmaxf(acc[ct][r] + bias, 0.0f));
        }
    }
    __syncthreads();

    // bilinear dot: thread -> (row, o); logits = left . M[b,o,:] + bil_b[o]
    {
        const int row = t >> 1;
        const int o   = t & 1;
        const int bb  = ctxid[row];
        const float* Mr = Mws + ((size_t)bb * ODIM + o) * HDIM;
        float d = 0.0f;
        #pragma unroll 8
        for (int i = 0; i < HDIM; ++i) d += bf2f(xn[row][i]) * Mr[i];
        lg[row][o] = d + bilB[o];
    }
    __syncthreads();

    // log_softmax over O=2 and store
    if (t < TB) {
        const int n = n0 + t;
        if (n < Nn) {
            const float l0 = lg[t][0], l1 = lg[t][1];
            const float mx = fmaxf(l0, l1);
            const float lse = mx + logf(expf(l0 - mx) + expf(l1 - mx));
            out_lp[(size_t)n * ODIM + 0] = l0 - lse;
            out_lp[(size_t)n * ODIM + 1] = l1 - lse;
        }
    }
}

// ---------------------------------------------------------------------------
// Launch
// ---------------------------------------------------------------------------
extern "C" void kernel_launch(void* const* d_in, const int* in_sizes, int n_in,
                              void* d_out, int out_size, void* d_ws, size_t ws_size,
                              hipStream_t stream) {
    const float* ctxv   = (const float*)d_in[0];   // [B,H]
    const float* X      = (const float*)d_in[1];   // [N,H]
    const float* Vmat   = (const float*)d_in[2];   // [C,H]
    const float* Kmat   = (const float*)d_in[3];   // [C,H]
    const float* leftW  = (const float*)d_in[4];   // [H,H]
    const float* leftB  = (const float*)d_in[5];   // [H]
    const float* rightW = (const float*)d_in[6];   // [H,H]
    const float* rightB = (const float*)d_in[7];   // [H]
    const float* bilW   = (const float*)d_in[8];   // [O,H,H]
    const float* bilB   = (const float*)d_in[9];   // [O]
    const int*   rc     = (const int*)d_in[10];    // [B]

    const int Bn = in_sizes[0] / HDIM;
    const int Nn = in_sizes[1] / HDIM;

    float* out    = (float*)d_out;
    float* out_lp = out;                            // [N,O]
    float* out_aw = out + (size_t)Nn * ODIM;        // [B,C]

    float* Mws  = (float*)d_ws;                                         // [B,O,H]
    int*   csum = (int*)((char*)d_ws + (size_t)Bn * ODIM * HDIM * sizeof(float));

    scan_kernel<<<1, 32, 0, stream>>>(rc, csum, Bn);
    attn_kernel<<<Bn, 256, 0, stream>>>(ctxv, Kmat, Vmat, rightW, rightB, bilW,
                                        Mws, out_aw);
    token_kernel<<<(Nn + TB - 1) / TB, 128, 0, stream>>>(X, leftW, leftB, bilB,
                                                         Mws, csum, Bn, Nn, out_lp);
}
